// CCALoss_1039382086287
// MI455X (gfx1250) — 8-bit matrix operand order repaired and checked by reading, compiled, not run on hardware
//
#include <hip/hip_runtime.h>
#include <hip/hip_bf16.h>
#include <math.h>

typedef __attribute__((ext_vector_type(8))) int v8i;

#define Bsz 4096
#define Cn  512
#define INV_T (1.0f / 0.07f)


__global__ void k_init(float* acc) {
    if (threadIdx.x < 8) acc[threadIdx.x] = 0.0f;
}

__global__ void __launch_bounds__(256) k_pack(const int* __restrict__ mc,
                                              unsigned char* __restrict__ pk,
                                              float* __restrict__ rs) {
    const int wave = threadIdx.x >> 5;
    const int lane = threadIdx.x & 31;
    const int row  = blockIdx.x * 8 + wave;
    const int* src = mc + (size_t)row * Cn + lane * 16;
    unsigned int uw[4] = {0,0,0,0}, vw[4] = {0,0,0,0};
    int cnt = 0;
#pragma unroll
    for (int k = 0; k < 16; ++k) {
        int c = src[k];
        unsigned int u = (c != 0) ? 1u : 0u;
        unsigned int v = (c == 1) ? 1u : 0u;
        uw[k >> 2] |= u << ((k & 3) * 8);
        vw[k >> 2] |= v << ((k & 3) * 8);
        cnt += (int)(u + v);
    }
    uint4 tu; tu.x = uw[0]; tu.y = uw[1]; tu.z = uw[2]; tu.w = uw[3];
    uint4 tv; tv.x = vw[0]; tv.y = vw[1]; tv.z = vw[2]; tv.w = vw[3];
    *(uint4*)(pk + (size_t)row * 1024 + lane * 16)       = tu;
    *(uint4*)(pk + (size_t)row * 1024 + 512 + lane * 16) = tv;
#pragma unroll
    for (int off = 16; off > 0; off >>= 1) cnt += __shfl_xor(cnt, off, 32);
    if (lane == 0) rs[row] = 0.5f * (float)cnt;
}

__global__ void __launch_bounds__(256) k_bce(const float* __restrict__ cl,
                                             const int* __restrict__ mc,
                                             float* __restrict__ acc) {
    float bsum = 0.f, msum = 0.f;
    const int n = Bsz * Cn;
    for (int i = blockIdx.x * 256 + threadIdx.x; i < n; i += gridDim.x * 256) {
        int c = mc[i];
        if (c != -1) {
            float x = cl[i];
            float sp = fmaxf(x, 0.f) + log1pf(expf(-fabsf(x)));
            bsum += sp - x * (float)c;
            msum += 1.f;
        }
    }
    __shared__ float sb[256], sm[256];
    sb[threadIdx.x] = bsum; sm[threadIdx.x] = msum;
    __syncthreads();
    for (int off = 128; off > 0; off >>= 1) {
        if (threadIdx.x < off) {
            sb[threadIdx.x] += sb[threadIdx.x + off];
            sm[threadIdx.x] += sm[threadIdx.x + off];
        }
        __syncthreads();
    }
    if (threadIdx.x == 0) {
        atomicAdd(&acc[2], sb[0]);
        atomicAdd(&acc[3], sm[0]);
    }
}

__global__ void __launch_bounds__(256) k_clip_lse(const float* __restrict__ img,
                                                  const float* __restrict__ txt,
                                                  float* __restrict__ acc) {
    const int bid = blockIdx.x;
    const int mat = bid >> 12;
    const int row = bid & (Bsz - 1);
    const float* r = (mat ? txt : img) + (size_t)row * Bsz;
    float m = -1e30f, s = 0.f;
    for (int j = threadIdx.x; j < Bsz; j += 256) {
        float x = r[j];
        if (x > m) { s *= expf(m - x); m = x; }
        s += expf(x - m);
    }
    __shared__ float smx[256], ssm[256];
    smx[threadIdx.x] = m; ssm[threadIdx.x] = s;
    __syncthreads();
    for (int off = 128; off > 0; off >>= 1) {
        if (threadIdx.x < off) {
            float ma = smx[threadIdx.x], sa = ssm[threadIdx.x];
            float mb = smx[threadIdx.x + off], sb = ssm[threadIdx.x + off];
            float mm = fmaxf(ma, mb);
            smx[threadIdx.x] = mm;
            ssm[threadIdx.x] = sa * expf(ma - mm) + sb * expf(mb - mm);
        }
        __syncthreads();
    }
    if (threadIdx.x == 0) {
        float lse = smx[0] + logf(ssm[0]);
        atomicAdd(&acc[mat], lse - r[row]);
    }
}

__global__ void __launch_bounds__(256) k_jaccard_kl(const unsigned char* __restrict__ pk,
                                                    const float* __restrict__ rs,
                                                    const float* __restrict__ cis,
                                                    float* __restrict__ acc) {
    const int m0   = blockIdx.x * 16;
    const int wave = threadIdx.x >> 5;
    const int lane = threadIdx.x & 31;
    const int half = lane >> 4;
    const int l15  = lane & 15;

    float rsm[8];
#pragma unroll
    for (int r = 0; r < 8; ++r) rsm[r] = rs[m0 + r + 8 * half];

    float mx[8], S[8], W[8], mx2[8], S2[8];
#pragma unroll
    for (int r = 0; r < 8; ++r) { mx[r] = -1e30f; S[r] = 0.f; W[r] = 0.f;
                                  mx2[r] = -1e30f; S2[r] = 0.f; }

    const unsigned char* apA = pk + (size_t)(m0 + l15) * 1024 + half * 16;

    for (int t = 0; t < 32; ++t) {
        const int n0 = (wave + 8 * t) * 16;
        const unsigned char* apB = pk + (size_t)(n0 + l15) * 1024 + half * 16;

        v8i c0 = {0,0,0,0,0,0,0,0};
        v8i c1 = {0,0,0,0,0,0,0,0};
#pragma unroll
        for (int kc = 0; kc < 16; ++kc) {
            const int kb = kc * 64;
            uint2 a01 = *(const uint2*)(apA + kb +  0);
            uint2 a23 = *(const uint2*)(apA + kb +  8);
            uint2 a45 = *(const uint2*)(apA + kb + 32);
            uint2 a67 = *(const uint2*)(apA + kb + 40);
            v8i a;
            a[0] = (int)a01.x; a[1] = (int)a01.y; a[2] = (int)a23.x; a[3] = (int)a23.y;
            a[4] = (int)a45.x; a[5] = (int)a45.y; a[6] = (int)a67.x; a[7] = (int)a67.y;
            uint4 b03 = *(const uint4*)(apB + kb +  0);
            uint4 b47 = *(const uint4*)(apB + kb + 32);
            v8i b;
            b[0] = (int)b03.x; b[1] = (int)b03.y; b[2] = (int)b03.z; b[3] = (int)b03.w;
            b[4] = (int)b47.x; b[5] = (int)b47.y; b[6] = (int)b47.z; b[7] = (int)b47.w;
            if (kc & 1)
                c1 = __builtin_amdgcn_wmma_i32_16x16x64_iu8(false, a, false, b, c1, false, false);
            else
                c0 = __builtin_amdgcn_wmma_i32_16x16x64_iu8(false, a, false, b, c0, false, false);
        }
        v8i cacc = c0 + c1;

        const int n = n0 + l15;
        const float rsn = rs[n];
        const float* ciscol = cis + n;
#pragma unroll
        for (int r = 0; r < 8; ++r) {
            const int m = m0 + r + 8 * half;
            float inter = 0.5f * (float)cacc[r];
            float uni   = rsm[r] + rsn - inter;
            float sim   = (uni > 0.f) ? (inter / uni) * INV_T : 0.f;
            float cv    = ciscol[(size_t)m * Bsz];
            if (sim > mx[r]) { float sc = expf(mx[r] - sim); S[r] *= sc; W[r] *= sc; mx[r] = sim; }
            float e = expf(sim - mx[r]);
            S[r] += e; W[r] += e * (sim - cv);
            if (cv > mx2[r]) { S2[r] *= expf(mx2[r] - cv); mx2[r] = cv; }
            S2[r] += expf(cv - mx2[r]);
        }
    }

    __shared__ float red[16][128][5];
    const int slot = wave * 16 + l15;
#pragma unroll
    for (int r = 0; r < 8; ++r) {
        float* p = red[r + 8 * half][slot];
        p[0] = mx[r]; p[1] = S[r]; p[2] = W[r]; p[3] = mx2[r]; p[4] = S2[r];
    }
    __syncthreads();
    if (threadIdx.x < 16) {
        const int row = threadIdx.x;
        float M = -1e30f, Ss = 0.f, Ww = 0.f, M2 = -1e30f, Q = 0.f;
        for (int s = 0; s < 128; ++s) {
            float* p = red[row][s];
            float mm = fmaxf(M, p[0]);
            float eA = expf(M - mm), eB = expf(p[0] - mm);
            Ss = Ss * eA + p[1] * eB;
            Ww = Ww * eA + p[2] * eB;
            M = mm;
            float mm2 = fmaxf(M2, p[3]);
            Q = Q * expf(M2 - mm2) + p[4] * expf(p[3] - mm2);
            M2 = mm2;
        }
        float kl = Ww / Ss - (M + logf(Ss)) + (M2 + logf(Q));
        atomicAdd(&acc[4], kl);
    }
}

__global__ void k_finalize(const float* __restrict__ acc, float* __restrict__ out) {
    float clip    = 0.5f * (acc[0] + acc[1]) / (float)Bsz;
    float concept = acc[2] / (acc[3] + 1e-8f);
    float kl      = acc[4] / (float)Bsz;
    out[0] = clip + 0.5f * concept + 0.3f * kl;
}

extern "C" void kernel_launch(void* const* d_in, const int* in_sizes, int n_in,
                              void* d_out, int out_size, void* d_ws, size_t ws_size,
                              hipStream_t stream) {
    const float* img = (const float*)d_in[0];
    const float* txt = (const float*)d_in[1];
    const float* cl  = (const float*)d_in[2];
    const float* cis = (const float*)d_in[3];
    const int*   mc  = (const int*)d_in[4];

    float*         acc = (float*)d_ws;
    float*         rsv = (float*)((char*)d_ws + 64);
    unsigned char* pk  = (unsigned char*)d_ws + 32768;

    k_init<<<1, 32, 0, stream>>>(acc);
    k_pack<<<Bsz / 8, 256, 0, stream>>>(mc, pk, rsv);
    k_bce<<<1024, 256, 0, stream>>>(cl, mc, acc);
    k_clip_lse<<<2 * Bsz, 256, 0, stream>>>(img, txt, acc);
    k_jaccard_kl<<<Bsz / 16, 256, 0, stream>>>(pk, rsv, cis, acc);
    k_finalize<<<1, 1, 0, stream>>>(acc, (float*)d_out);
}
